// CognateLTMBank_61194694034001
// MI455X (gfx1250) — matrix operand order repaired and checked by reading, not compiled, not run on hardware
//
#include <hip/hip_runtime.h>
#include <hip/hip_bf16.h>
#include <math.h>

typedef __attribute__((ext_vector_type(16))) _Float16 v16h;
typedef __attribute__((ext_vector_type(8)))  float    v8f;

#define B_SZ   1024
#define N_SZ   65536
#define D_SZ   256
#define TOPK   32
#define SIM_TH 0.1f
#define EPSF   1e-8f

#define SPLITS 64
#define KPB    (N_SZ / SPLITS)          // 1024 keys per block
#define TM     32                       // query rows per block (2 WMMA M-tiles)
#define WAVES  8
#define TPW    (KPB / 16 / WAVES)       // 8 sixteen-wide key tiles per wave
#define CAND   (SPLITS * TOPK)          // 2048 candidates per row

// ---------------- row normalization: f32 row -> f16 normalized row ----------
__global__ __launch_bounds__(256) void normalize_rows_f16(
    const float* __restrict__ src, _Float16* __restrict__ dst) {
  const int row = blockIdx.x;
  const int tid = threadIdx.x;            // 256 threads == D
  float v = src[(size_t)row * D_SZ + tid];
  float ss = v * v;
  #pragma unroll
  for (int off = 16; off > 0; off >>= 1) ss += __shfl_xor(ss, off, 32);
  __shared__ float wss[WAVES];
  const int w = tid >> 5, lane = tid & 31;
  if (lane == 0) wss[w] = ss;
  __syncthreads();
  float tot = 0.f;
  #pragma unroll
  for (int i = 0; i < WAVES; ++i) tot += wss[i];
  const float norm = fmaxf(sqrtf(tot), EPSF);
  dst[(size_t)row * D_SZ + tid] = (_Float16)(v / norm);
}

// --- fused WMMA sims + per-split top-32 (32 query rows x 1024 keys/block) ---
// Each wave: one 16-key B tile feeds TWO WMMAs (M-tiles 0 and 1) -> 2x the
// FLOP per B byte vs single-M blocking; keys stay L2-resident (32MB < 192MB).
__global__ __launch_bounds__(256) void sims_topk_partial(
    const _Float16* __restrict__ qn, const _Float16* __restrict__ kn,
    float* __restrict__ cand_val, int* __restrict__ cand_idx) {
  __shared__ _Float16 qs[TM * D_SZ];     // 16 KB
  __shared__ float    sims[TM * KPB];    // 128 KB (total 144 KB < 320 KB WGP)

  const int m0    = blockIdx.x * TM;
  const int split = blockIdx.y;
  const int nbase = split * KPB;
  const int tid   = threadIdx.x;
  const int wave  = tid >> 5;
  const int lane  = tid & 31;
  const int r     = lane & 15;           // A row (within M-tile) / B col / D col
  const int hi    = lane >> 4;

  for (int i = tid; i < TM * D_SZ; i += 256) qs[i] = qn[(size_t)m0 * D_SZ + i];
  __syncthreads();

  for (int t = 0; t < TPW; ++t) {
    const int nb = (wave * TPW + t) * 16;
    const _Float16* krow = kn + (size_t)(nbase + nb + r) * D_SZ;
    if (t + 1 < TPW)  // warm next key tile (global_prefetch_b8)
      __builtin_prefetch(kn + (size_t)(nbase + nb + 16 + r) * D_SZ, 0, 1);

    v8f acc0 = {}, acc1 = {};
    #pragma unroll
    for (int kk = 0; kk < D_SZ; kk += 32) {
      v16h a0, a1, b;
      #pragma unroll
      for (int j = 0; j < 8; ++j) {
        const int ka = (j < 4) ? (hi * 8 + 2 * j) : (16 + hi * 8 + 2 * (j - 4));
        a0[2 * j]     = qs[r * D_SZ + kk + ka];
        a0[2 * j + 1] = qs[r * D_SZ + kk + ka + 1];
        a1[2 * j]     = qs[(16 + r) * D_SZ + kk + ka];
        a1[2 * j + 1] = qs[(16 + r) * D_SZ + kk + ka + 1];
        const int kb = ka;  // B: same K order as A
        b[2 * j]     = krow[kk + kb];
        b[2 * j + 1] = krow[kk + kb + 1];
      }
      acc0 = __builtin_amdgcn_wmma_f32_16x16x32_f16(false, a0, false, b,
                                                    (short)0, acc0, false, false);
      acc1 = __builtin_amdgcn_wmma_f32_16x16x32_f16(false, a1, false, b,
                                                    (short)0, acc1, false, false);
    }
    #pragma unroll
    for (int v = 0; v < 8; ++v) {
      const int m = hi * 8 + v;
      sims[m * KPB + nb + r]          = acc0[v];
      sims[(16 + m) * KPB + nb + r]   = acc1[v];
    }
  }
  __syncthreads();

  // each wave selects top-32 for TM/WAVES = 4 rows; owner-lane invalidation
  for (int rr = 0; rr < TM / WAVES; ++rr) {
    const int row = wave * (TM / WAVES) + rr;
    float* srow = sims + row * KPB;
    for (int sel = 0; sel < TOPK; ++sel) {
      float best = -INFINITY; int bestc = KPB;
      for (int i = lane; i < KPB; i += 32) {
        const float x = srow[i];
        if (x > best || (x == best && i < bestc)) { best = x; bestc = i; }
      }
      #pragma unroll
      for (int off = 16; off > 0; off >>= 1) {
        const float ov = __shfl_xor(best, off, 32);
        const int   oc = __shfl_xor(bestc, off, 32);
        if (ov > best || (ov == best && oc < bestc)) { best = ov; bestc = oc; }
      }
      if (lane == 0) {
        const size_t o = ((size_t)(m0 + row) * SPLITS + split) * TOPK + sel;
        cand_val[o] = best;
        cand_idx[o] = nbase + bestc;
      }
      if (bestc < KPB && (bestc & 31) == lane) srow[bestc] = -INFINITY;
    }
  }
}

// ---------------- merge 64x32 candidates -> final top-32 per row ------------
__global__ __launch_bounds__(32) void merge_topk(
    const float* __restrict__ cand_val, const int* __restrict__ cand_idx,
    float* __restrict__ out_sims, int* __restrict__ top_idx) {
  __shared__ float v[CAND];
  __shared__ int   ix[CAND];
  const int b = blockIdx.x, lane = threadIdx.x;
  for (int i = lane; i < CAND; i += 32) {
    v[i]  = cand_val[(size_t)b * CAND + i];
    ix[i] = cand_idx[(size_t)b * CAND + i];
  }
  __syncthreads();
  for (int sel = 0; sel < TOPK; ++sel) {
    float best = -INFINITY; int bestp = CAND;
    for (int i = lane; i < CAND; i += 32) {
      const float x = v[i];
      if (x > best || (x == best && i < bestp)) { best = x; bestp = i; }
    }
    #pragma unroll
    for (int off = 16; off > 0; off >>= 1) {
      const float ov = __shfl_xor(best, off, 32);
      const int   op = __shfl_xor(bestp, off, 32);
      if (ov > best || (ov == best && op < bestp)) { best = ov; bestp = op; }
    }
    if (lane == 0) {
      const bool valid = best >= SIM_TH;
      out_sims[(size_t)b * TOPK + sel] = valid ? best : 0.0f;
      top_idx[(size_t)b * TOPK + sel]  = valid ? ix[bestp] : -1;
    }
    if (bestp < CAND && (bestp & 31) == lane) v[bestp] = -INFINITY;
  }
}

// ---------------- gather values/keys (zero-pad invalid) ---------------------
__global__ __launch_bounds__(64) void gather_vk(
    const float* __restrict__ mvals, const float* __restrict__ mkeys,
    const int* __restrict__ top_idx,
    float* __restrict__ out_vals, float* __restrict__ out_keys) {
  const int bk = blockIdx.x, t = threadIdx.x;   // 64 threads x float4 = 256
  const int idx = top_idx[bk];
  float4 vv = {0.f, 0.f, 0.f, 0.f}, kv = {0.f, 0.f, 0.f, 0.f};
  if (idx >= 0) {
    vv = ((const float4*)(mvals + (size_t)idx * D_SZ))[t];
    kv = ((const float4*)(mkeys + (size_t)idx * D_SZ))[t];
  }
  ((float4*)(out_vals + (size_t)bk * D_SZ))[t] = vv;
  ((float4*)(out_keys + (size_t)bk * D_SZ))[t] = kv;
}

// ---------------- usage: copy then scatter same-value boost -----------------
__global__ void usage_copy(const float* __restrict__ usage,
                           float* __restrict__ out_usage) {
  const int i = blockIdx.x * blockDim.x + threadIdx.x;
  if (i < N_SZ) out_usage[i] = usage[i];
}
__global__ void usage_update(const float* __restrict__ usage,
                             const int* __restrict__ top_idx,
                             float* __restrict__ out_usage) {
  const int i = blockIdx.x * blockDim.x + threadIdx.x;
  if (i < B_SZ * TOPK) {
    const int idx = top_idx[i];
    if (idx >= 0) out_usage[idx] = fminf(usage[idx] * 1.1f, 2.0f);
  }
}

extern "C" void kernel_launch(void* const* d_in, const int* in_sizes, int n_in,
                              void* d_out, int out_size, void* d_ws, size_t ws_size,
                              hipStream_t stream) {
  (void)in_sizes; (void)n_in; (void)out_size; (void)ws_size;
  const float* query  = (const float*)d_in[0];   // [B, D]
  const float* mkeys  = (const float*)d_in[1];   // [N, D]
  const float* mvals  = (const float*)d_in[2];   // [N, D]
  const float* musage = (const float*)d_in[3];   // [N]

  // output layout: values [B*K*D] | keys [B*K*D] | sims [B*K] | usage [N]
  float* out_vals  = (float*)d_out;
  float* out_keys  = out_vals + (size_t)B_SZ * TOPK * D_SZ;
  float* out_sims  = out_keys + (size_t)B_SZ * TOPK * D_SZ;
  float* out_usage = out_sims + (size_t)B_SZ * TOPK;

  // workspace layout (~48.7 MB)
  char* w = (char*)d_ws;
  _Float16* qn       = (_Float16*)w; w += (size_t)B_SZ * D_SZ * sizeof(_Float16);
  _Float16* kn       = (_Float16*)w; w += (size_t)N_SZ * D_SZ * sizeof(_Float16);
  float*    cand_val = (float*)w;    w += (size_t)B_SZ * SPLITS * TOPK * sizeof(float);
  int*      cand_idx = (int*)w;      w += (size_t)B_SZ * SPLITS * TOPK * sizeof(int);
  int*      top_idx  = (int*)w;

  normalize_rows_f16<<<B_SZ, 256, 0, stream>>>(query, qn);
  normalize_rows_f16<<<N_SZ, 256, 0, stream>>>(mkeys, kn);

  sims_topk_partial<<<dim3(B_SZ / TM, SPLITS), 256, 0, stream>>>(
      qn, kn, cand_val, cand_idx);

  merge_topk<<<B_SZ, 32, 0, stream>>>(cand_val, cand_idx, out_sims, top_idx);

  gather_vk<<<B_SZ * TOPK, 64, 0, stream>>>(mvals, mkeys, top_idx,
                                            out_vals, out_keys);

  usage_copy<<<N_SZ / 256, 256, 0, stream>>>(musage, out_usage);
  usage_update<<<(B_SZ * TOPK) / 256, 256, 0, stream>>>(musage, top_idx,
                                                        out_usage);
}
